// Attention_39865886442227
// MI455X (gfx1250) — matrix operand order repaired and checked by reading, not compiled, not run on hardware
//
#include <hip/hip_runtime.h>
#include <hip/hip_bf16.h>

// ---------------------------------------------------------------------------
// MI455X (gfx1250, wave32) local-window attention, bf16 WMMA pipeline.
//   K0: one-time f32->bf16 convert of w_qkv / w_proj (removes converts from
//       the GEMM inner loops; weights live in L2 as bf16)
//   K1: qkv = x @ w_qkv.T          (bf16 WMMA, LDS-staged A)
//   K2: band flash-attention        (2x S-WMMA + 2x PV-WMMA per 32-key chunk)
//   K3: y = attn @ w_proj.T + b     (bf16 WMMA)
// f32->bf16 uses packed hardware converts (v_cvt_pk_bf16_f32).
// ---------------------------------------------------------------------------

typedef __attribute__((ext_vector_type(16))) __bf16 v16bf;
typedef __attribute__((ext_vector_type(8)))  __bf16 v8bf;
typedef __attribute__((ext_vector_type(4)))  __bf16 v4bf;
typedef __attribute__((ext_vector_type(2)))  __bf16 v2bf;
typedef __attribute__((ext_vector_type(8)))  float  v8f;
typedef __attribute__((ext_vector_type(4)))  float  v4f;

union BF16x8 { v8bf v; v2bf p[4]; };
union BF16x4 { v4bf v; v2bf p[2]; };

// Packed f32 pair -> bf16 pair (hardware v_cvt_pk_bf16_f32 when declared).
__device__ __forceinline__ v2bf f2bf2(float a, float b) {
#if __has_builtin(__builtin_amdgcn_cvt_pk_bf16_f32)
  return __builtin_amdgcn_cvt_pk_bf16_f32(a, b);
#else
  v2bf r; r[0] = (__bf16)a; r[1] = (__bf16)b; return r;
#endif
}

__device__ __forceinline__ v16bf cat8(v8bf lo, v8bf hi) {
  return __builtin_shufflevector(lo, hi, 0,1,2,3,4,5,6,7,8,9,10,11,12,13,14,15);
}

// Load a 16-element bf16 fragment from two contiguous b128s.
__device__ __forceinline__ v16bf load_frag(const __bf16* __restrict__ p) {
  return cat8(*(const v8bf*)p, *(const v8bf*)(p + 8));
}

__device__ __forceinline__ v8f wmma_bf16(v16bf a, v16bf b, v8f c) {
  // (neg_a, A, neg_b, B, c_mod, C, reuse_a, reuse_b)
  return __builtin_amdgcn_wmma_f32_16x16x32_bf16(false, a, false, b, (short)0, c,
                                                 false, false);
}

__device__ __forceinline__ float rmax16(float v) {
  v = fmaxf(v, __shfl_xor(v, 1, 32));
  v = fmaxf(v, __shfl_xor(v, 2, 32));
  v = fmaxf(v, __shfl_xor(v, 4, 32));
  v = fmaxf(v, __shfl_xor(v, 8, 32));
  return v;
}
__device__ __forceinline__ float rsum16(float v) {
  v += __shfl_xor(v, 1, 32);
  v += __shfl_xor(v, 2, 32);
  v += __shfl_xor(v, 4, 32);
  v += __shfl_xor(v, 8, 32);
  return v;
}

// ------------------- Kernel 0: one-time weight conversion -------------------
__global__ __launch_bounds__(256)
void convert_w_kernel(const float* __restrict__ src, __bf16* __restrict__ dst,
                      int n4)
{
  const int i = blockIdx.x * 256 + threadIdx.x;
  if (i < n4) {
    v4f f = ((const v4f*)src)[i];
    BF16x4 r;
    r.p[0] = f2bf2(f.x, f.y);
    r.p[1] = f2bf2(f.z, f.w);
    ((v4bf*)dst)[i] = r.v;
  }
}

// ------------------------- Kernel 1: QKV projection -------------------------
// grid.x = 2048 row-tiles of 16 tokens, block = 256 (8 waves).
// Output layouts:  q,k : [B,h,N,32] bf16 ;  v : [B,h,32,N] bf16 (transposed)
__global__ __launch_bounds__(256)
void qkv_proj_kernel(const float* __restrict__ x, const __bf16* __restrict__ wqkv,
                     __bf16* __restrict__ qb, __bf16* __restrict__ kbuf,
                     __bf16* __restrict__ vtb)
{
  __shared__ __align__(16) __bf16 lds_a[16 * 256];
  const int rowbase = blockIdx.x * 16;
  const int t = threadIdx.x;

  { // stage x row-tile (16x256 f32) into LDS as bf16, two ds_store_b128/thread
    const int m  = t >> 4;
    const int k0 = (t & 15) * 16;
    const v4f* src = (const v4f*)(x + (size_t)(rowbase + m) * 256 + k0);
    BF16x8 lo, hi;
#pragma unroll
    for (int i = 0; i < 2; ++i) {
      v4f f = src[i];
      lo.p[i*2+0] = f2bf2(f.x, f.y);
      lo.p[i*2+1] = f2bf2(f.z, f.w);
    }
#pragma unroll
    for (int i = 0; i < 2; ++i) {
      v4f f = src[2 + i];
      hi.p[i*2+0] = f2bf2(f.x, f.y);
      hi.p[i*2+1] = f2bf2(f.z, f.w);
    }
    *(v8bf*)&lds_a[m * 256 + k0]     = lo.v;
    *(v8bf*)&lds_a[m * 256 + k0 + 8] = hi.v;
  }
  __syncthreads();

  const int wave = t >> 5, lane = t & 31;
  const int l16 = lane & 15, h2 = lane >> 4;
  const int bidx = rowbase >> 10, tokbase = rowbase & 1023;

#pragma unroll 1
  for (int ci = 0; ci < 6; ++ci) {           // 48 col-tiles / 8 waves
    const int colbase = (wave * 6 + ci) * 16;
    v8f acc = {};
#pragma unroll
    for (int kc = 0; kc < 8; ++kc) {
      const int kbase = kc * 32;
      v16bf A = cat8(*(const v8bf*)&lds_a[l16 * 256 + kbase + 16 * h2],
                     *(const v8bf*)&lds_a[l16 * 256 + kbase + 16 * h2 + 8]);
      v16bf B = load_frag(wqkv + (size_t)(colbase + l16) * 256 + kbase + 16 * h2);
      acc = wmma_bf16(A, B, acc);
    }
    const int which = colbase >> 8;          // 0=q 1=k 2=v
    const int head  = (colbase >> 5) & 7;
    const int ddb   = colbase & 31;          // 0 or 16
    if (which == 2) {                        // v: transposed store, contiguous b128
      BF16x8 pk;
#pragma unroll
      for (int j = 0; j < 4; ++j) pk.p[j] = f2bf2(acc[2*j], acc[2*j+1]);
      size_t off = ((size_t)(bidx * 8 + head) * 32 + ddb + l16) * 1024
                 + (size_t)tokbase + 8 * h2;
      *(v8bf*)(vtb + off) = pk.v;
    } else {
      __bf16* dst = (which == 0) ? qb : kbuf;
      const size_t base = (size_t)(bidx * 8 + head) * 1024;
#pragma unroll
      for (int j = 0; j < 4; ++j) {
        v2bf p = f2bf2(acc[2*j], acc[2*j+1]);
        const int tok = tokbase + 2*j + 8 * h2;
        dst[(base + tok)     * 32 + ddb + l16] = p[0];
        dst[(base + tok + 1) * 32 + ddb + l16] = p[1];
      }
    }
  }
}

// ------------------ Kernel 2: banded flash attention ------------------------
// grid = (64 q-tiles, 8 heads, 32 batch), block = 32 (one wave per q-tile).
__global__ __launch_bounds__(32)
void attn_kernel(const __bf16* __restrict__ qb, const __bf16* __restrict__ kbuf,
                 const __bf16* __restrict__ vtb, __bf16* __restrict__ attnb)
{
  __shared__ __align__(16) __bf16 lds_p[16 * 32];
  const int qt = blockIdx.x, head = blockIdx.y, b = blockIdx.z;
  const int qbase = qt * 16;
  const int lane  = threadIdx.x;
  const int l16 = lane & 15, h2 = lane >> 4;
  const float scale = 0.17677669529663687f;   // 32^-0.5

  const size_t bh = (size_t)(b * 8 + head);
  const __bf16* qptr = qb   + bh * 1024 * 32;
  const __bf16* kptr = kbuf + bh * 1024 * 32;
  const __bf16* vptr = vtb  + bh * 32 * 1024;

  // Q A-fragment (row = query l16, one contiguous 16-elem K-run per lane half)
  v16bf QA = cat8(*(const v8bf*)(qptr + (size_t)(qbase + l16) * 32 + 16 * h2),
                  *(const v8bf*)(qptr + (size_t)(qbase + l16) * 32 + 16 * h2 + 8));

  float mrow[8], lrow[8];
  int qh[8], qw[8];
  v8f o0 = {}, o1 = {};
#pragma unroll
  for (int j = 0; j < 8; ++j) {
    mrow[j] = -1e30f; lrow[j] = 0.f;
    const int qtok = qbase + j + 8 * h2;
    qh[j] = qtok >> 6; qw[j] = qtok & 63;
  }

  // window (7,11) on 16x64 grid -> flat band q-197 .. q+15+197
  int klo = qbase - 197; if (klo < 0) klo = 0;
  int khi = qbase + 15 + 197; if (khi > 1023) khi = 1023;

#pragma unroll 1
  for (int kc = klo & ~31; kc <= khi; kc += 32) {
    if (kc + 32 <= khi)
      __builtin_prefetch(kptr + (size_t)(kc + 32) * 32, 0, 1);  // global_prefetch_b8

    // K B-fragments for the two 16-key tiles of this chunk
    v16bf KB0 = load_frag(kptr + (size_t)(kc + l16) * 32 + 16 * h2);
    v16bf KB1 = load_frag(kptr + (size_t)(kc + 16 + l16) * 32 + 16 * h2);

    v8f zero = {};
    v8f s0 = wmma_bf16(QA, KB0, zero);
    v8f s1 = wmma_bf16(QA, KB1, zero);

    const int kt0 = kc + l16, kt1 = kc + 16 + l16;
    const int kh0 = kt0 >> 6, kw0 = kt0 & 63;
    const int kh1 = kt1 >> 6, kw1 = kt1 & 63;
    float corr[8];
#pragma unroll
    for (int j = 0; j < 8; ++j) {
      int d0h = qh[j] - kh0; d0h = d0h < 0 ? -d0h : d0h;
      int d0w = qw[j] - kw0; d0w = d0w < 0 ? -d0w : d0w;
      int d1h = qh[j] - kh1; d1h = d1h < 0 ? -d1h : d1h;
      int d1w = qw[j] - kw1; d1w = d1w < 0 ? -d1w : d1w;
      const bool v0 = (d0h <= 3) && (d0w <= 5);
      const bool v1 = (d1h <= 3) && (d1w <= 5);
      float a0 = v0 ? s0[j] * scale : -1e30f;
      float a1 = v1 ? s1[j] * scale : -1e30f;
      const float rm = rmax16(fmaxf(a0, a1));
      const float mn = fmaxf(mrow[j], rm);
      const float c  = __expf(mrow[j] - mn);
      const float p0 = v0 ? __expf(a0 - mn) : 0.f;
      const float p1 = v1 ? __expf(a1 - mn) : 0.f;
      lrow[j] = lrow[j] * c + rsum16(p0 + p1);
      mrow[j] = mn;
      corr[j] = c;
      v2bf pp = f2bf2(p0, p1);                         // one pk convert per row
      lds_p[(j + 8 * h2) * 32 + l16]      = pp[0];     // C-layout -> LDS
      lds_p[(j + 8 * h2) * 32 + 16 + l16] = pp[1];
    }
#pragma unroll
    for (int j = 0; j < 8; ++j) { o0[j] *= corr[j]; o1[j] *= corr[j]; }
    __syncthreads();

    // P as A-fragment from LDS (relayout), V B-fragments contiguous (transposed V)
    v16bf PA  = cat8(*(const v8bf*)&lds_p[l16 * 32 + 16 * h2],
                     *(const v8bf*)&lds_p[l16 * 32 + 16 * h2 + 8]);
    v16bf VB0 = load_frag(vptr + (size_t)l16 * 1024 + kc + 16 * h2);
    v16bf VB1 = load_frag(vptr + (size_t)(16 + l16) * 1024 + kc + 16 * h2);
    o0 = wmma_bf16(PA, VB0, o0);
    o1 = wmma_bf16(PA, VB1, o1);
    __syncthreads();
  }

#pragma unroll
  for (int j = 0; j < 8; ++j) {
    const float inv = lrow[j] > 0.f ? 1.0f / lrow[j] : 0.f;
    const int tok = qbase + j + 8 * h2;
    const size_t rb = ((size_t)b * 1024 + tok) * 256 + head * 32;
    v2bf oo = f2bf2(o0[j] * inv, o1[j] * inv);
    attnb[rb + l16]      = oo[0];
    attnb[rb + 16 + l16] = oo[1];
  }
}

// ---------------------- Kernel 3: output projection -------------------------
// grid.x = 2048 row-tiles, block = 256 (8 waves x 2 col-tiles).
__global__ __launch_bounds__(256)
void proj_kernel(const __bf16* __restrict__ attnb, const __bf16* __restrict__ wproj,
                 const float* __restrict__ bias, float* __restrict__ out)
{
  const int rowbase = blockIdx.x * 16;
  const int t = threadIdx.x;
  const int wave = t >> 5, lane = t & 31;
  const int l16 = lane & 15, h2 = lane >> 4;

#pragma unroll 1
  for (int ci = 0; ci < 2; ++ci) {
    const int colbase = (wave * 2 + ci) * 16;
    v8f acc = {};
#pragma unroll
    for (int kc = 0; kc < 8; ++kc) {
      const int kbase = kc * 32;
      const __bf16* ap = attnb + (size_t)(rowbase + l16) * 256 + kbase;
      v16bf A = cat8(*(const v8bf*)(ap + 16 * h2), *(const v8bf*)(ap + 16 * h2 + 8));
      v16bf B = load_frag(wproj + (size_t)(colbase + l16) * 256 + kbase + 16 * h2);
      acc = wmma_bf16(A, B, acc);
    }
    const float bv = bias[colbase + l16];
#pragma unroll
    for (int j = 0; j < 8; ++j)
      out[(size_t)(rowbase + j + 8 * h2) * 256 + colbase + l16] = acc[j] + bv;
  }
}

// ---------------------------------------------------------------------------
extern "C" void kernel_launch(void* const* d_in, const int* in_sizes, int n_in,
                              void* d_out, int out_size, void* d_ws, size_t ws_size,
                              hipStream_t stream) {
  const float* x     = (const float*)d_in[0];
  const float* wqkv  = (const float*)d_in[1];
  const float* wproj = (const float*)d_in[2];
  const float* bproj = (const float*)d_in[3];
  // d_in[4] (mask) unused: the 7x11 window is evaluated analytically.
  float* out = (float*)d_out;

  const size_t QSZ  = (size_t)32 * 8 * 1024 * 32;    // 8M bf16 = 16 MB each
  const int WQKV_N  = 768 * 256;                     // 196608 f32
  const int WPROJ_N = 256 * 256;                     //  65536 f32
  __bf16* qb       = (__bf16*)d_ws;
  __bf16* kbuf     = qb    + QSZ;
  __bf16* vtb      = kbuf  + QSZ;
  __bf16* attnb    = vtb   + QSZ;
  __bf16* wqkv_bf  = attnb + QSZ;                    // 384 KB
  __bf16* wproj_bf = wqkv_bf + WQKV_N;               // 128 KB   (total ~64.5 MB)

  convert_w_kernel<<<WQKV_N / 4 / 256, 256, 0, stream>>>(wqkv, wqkv_bf, WQKV_N / 4);
  convert_w_kernel<<<WPROJ_N / 4 / 256, 256, 0, stream>>>(wproj, wproj_bf, WPROJ_N / 4);
  qkv_proj_kernel<<<2048, 256, 0, stream>>>(x, wqkv_bf, qb, kbuf, vtb);
  attn_kernel<<<dim3(64, 8, 32), 32, 0, stream>>>(qb, kbuf, vtb, attnb);
  proj_kernel<<<2048, 256, 0, stream>>>(attnb, wproj_bf, bproj, out);
}
